// CenteringAttention_41558103556647
// MI455X (gfx1250) — matrix operand order repaired and checked by reading, not compiled, not run on hardware
//
#include <hip/hip_runtime.h>
#include <hip/hip_bf16.h>
#include <math.h>

// ---------------------------------------------------------------------------
// CenteringAttention for MI455X (gfx1250): flash-attention fused, bf16 WMMA.
//   B=4, C=256, H=W=64 -> N=4096 tokens, Cq=32.
//   attn_kernel: flash attention; K/V tiles double-buffered via async
//                global->LDS copies overlapped with WMMA compute.
//   proj_kernel: Q/K token-major [t][32], V channel-major [c][t] (bf16).
// ---------------------------------------------------------------------------

typedef __bf16 bf16_t;
typedef __attribute__((ext_vector_type(16))) __bf16 v16bf;
typedef __attribute__((ext_vector_type(8)))  __bf16 v8bf;
typedef __attribute__((ext_vector_type(4)))  __bf16 v4bf;
typedef __attribute__((ext_vector_type(2)))  __bf16 v2bf;
typedef __attribute__((ext_vector_type(8)))  float  v8f;

#define AS1 __attribute__((address_space(1)))
#define AS3 __attribute__((address_space(3)))
typedef int i32x4 __attribute__((vector_size(16)));

#define C_DIM 256
#define N_TOK 4096
#define CQ    32

static __device__ __forceinline__ v16bf cat8(v8bf lo, v8bf hi) {
  return __builtin_shufflevector(lo, hi, 0,1,2,3,4,5,6,7,8,9,10,11,12,13,14,15);
}

// ---- async global->LDS 16-byte copy (CDNA5 GLOBAL_LOAD_ASYNC_TO_LDS_B128),
// ---- gated so the source compiles on toolchains without the builtin.
#if __has_builtin(__builtin_amdgcn_global_load_async_to_lds_b128)
#define HAVE_ASYNC_COPY 1
static __device__ __forceinline__ void copy16_g2l(const bf16_t* g, bf16_t* l) {
  __builtin_amdgcn_global_load_async_to_lds_b128(
      (AS1 i32x4*)g, (AS3 i32x4*)l, 0, 0);
}
#else
#define HAVE_ASYNC_COPY 0
static __device__ __forceinline__ void copy16_g2l(const bf16_t* g, bf16_t* l) {
  *(v8bf*)l = *(const v8bf*)g;
}
#endif

static __device__ __forceinline__ void async_wait_all() {
#if HAVE_ASYNC_COPY
#if __has_builtin(__builtin_amdgcn_s_wait_asynccnt)
  __builtin_amdgcn_s_wait_asynccnt(0);
#endif
#endif
}

// ===========================================================================
// Fused flash attention + residual.
// Block: 256 thr (8 waves), 64 query tokens per block, grid = (N/64, B).
// KV tiles of 32 double-buffered: async copies for tile jt+1 issued before
// computing tile jt, so the DMA overlaps WMMA + softmax work.
// O[256c x 64i] lives in VGPRs (wave w owns c in [32w, 32w+32)).
// ===========================================================================
__global__ __launch_bounds__(256)
void attn_kernel(const float* __restrict__ x,
                 const bf16_t* __restrict__ Qb, const bf16_t* __restrict__ Kb,
                 const bf16_t* __restrict__ Vb, float* __restrict__ out) {
  __shared__ bf16_t sQ[64][48];        // [i][d]  96B stride
  __shared__ bf16_t sK[2][32][48];     // [buf][j][d]
  __shared__ bf16_t sV[2][256][40];    // [buf][c][j]  channel-major Vb
  __shared__ float  sS[64][36];        // scores [i][j]
  __shared__ bf16_t sP[64][48];        // exp probs [i][j]
  __shared__ float  sM[64], sL[64], sScale[64];

  const int tid  = threadIdx.x;
  const int lane = tid & 31;
  const int wave = tid >> 5;
  const int b    = blockIdx.y;
  const int i0   = blockIdx.x * 64;

  const bf16_t* Qg = Qb + (size_t)b * N_TOK * CQ;
  const bf16_t* Kg = Kb + (size_t)b * N_TOK * CQ;
  const bf16_t* Vg = Vb + (size_t)b * N_TOK * C_DIM;   // [c][t]

  // ---- prologue: resident Q tile + KV tile 0 into buffer 0 ----
  {
    int e = tid * 8;                  // 2048 bf16
    int t = e >> 5, dd = e & 31;
    copy16_g2l(&Qg[(size_t)(i0 + t)*CQ + dd], &sQ[t][dd]);
  }
  if (tid < 128) {
    int e = tid * 8; int t = e >> 5, dd = e & 31;
    copy16_g2l(&Kg[(size_t)t*CQ + dd], &sK[0][t][dd]);
  }
  #pragma unroll
  for (int u = 0; u < 4; ++u) {
    int e = (tid + u*256) * 8;        // 8192 bf16
    int c = e >> 5, jj = e & 31;
    copy16_g2l(&Vg[(size_t)c*N_TOK + jj], &sV[0][c][jj]);
  }
  if (tid < 64) { sM[tid] = -1e30f; sL[tid] = 0.0f; }

  v8f o[8];
  v8f vzero = {};
  #pragma unroll
  for (int i = 0; i < 8; ++i) o[i] = vzero;

  const int itS  = wave & 3;     // S tile: query 16-block
  const int jtS  = wave >> 2;    // S tile: key 16-block (0..1)
  const int cblk = wave * 32;    // PV: this wave's 32 output channels

  for (int jt = 0; jt < N_TOK/32; ++jt) {
    // current buffer's copies (issued last iteration) complete + visible
    async_wait_all();
    __syncthreads();

    const int cur = jt & 1;

    if (jt + 2 < N_TOK/32) {     // pull tile jt+2 toward L2
      __builtin_prefetch(&Vg[(size_t)(tid & 255)*N_TOK + jt*32 + 64], 0, 1);
    }
    // ---- issue async copies for tile jt+1 into the other buffer ----
    if (jt + 1 < N_TOK/32) {
      const int nj0 = (jt + 1) * 32;
      const int nb  = cur ^ 1;
      if (tid < 128) {
        int e = tid * 8; int t = e >> 5, dd = e & 31;
        copy16_g2l(&Kg[(size_t)(nj0 + t)*CQ + dd], &sK[nb][t][dd]);
      }
      #pragma unroll
      for (int u = 0; u < 4; ++u) {
        int e = (tid + u*256) * 8;
        int c = e >> 5, jj = e & 31;
        copy16_g2l(&Vg[(size_t)c*N_TOK + nj0 + jj], &sV[nb][c][jj]);
      }
    }

    // ---- S = Q * K^T : one 16x16 tile per wave (K-dim 32 = one WMMA) ----
    {
      int row = itS*16 + (lane & 15);
      int klo = (lane < 16) ? 0 : 16;
      v16bf qf = cat8(*(const v8bf*)&sQ[row][klo],
                      *(const v8bf*)&sQ[row][klo + 8]);
      int col  = jtS*16 + (lane & 15);
      int doff = (lane < 16) ? 0 : 16;
      v16bf kf = *(const v16bf*)&sK[cur][col][doff];  // B[k=d][n=j] = K[j][d]
      v8f s = {};
      s = __builtin_amdgcn_wmma_f32_16x16x32_bf16(false, qf, false, kf,
                                                  (short)0, s, false, false);
      int nn = lane & 15, mb = 8*(lane >> 4);
      #pragma unroll
      for (int r = 0; r < 8; ++r)
        sS[itS*16 + mb + r][jtS*16 + nn] = s[r];
    }
    __syncthreads();

    // ---- online softmax over this 32-key tile (thread per query row) ----
    if (tid < 64) {
      float mOld = sM[tid], mNew = mOld;
      #pragma unroll 8
      for (int j = 0; j < 32; ++j) mNew = fmaxf(mNew, sS[tid][j]);
      float scale = __expf(mOld - mNew);
      float psum = 0.0f;
      #pragma unroll 4
      for (int j = 0; j < 32; j += 2) {      // packed bf16-pair stores
        float p0 = __expf(sS[tid][j+0] - mNew);
        float p1 = __expf(sS[tid][j+1] - mNew);
        v2bf pp; pp[0] = (bf16_t)p0; pp[1] = (bf16_t)p1;
        *(v2bf*)&sP[tid][j] = pp;
        psum += p0 + p1;
      }
      sL[tid] = sL[tid]*scale + psum;
      sM[tid] = mNew;
      sScale[tid] = scale;
    }
    __syncthreads();

    // ---- O[c,i] = O*scale_i + V^T * P^T  (8 WMMAs per wave) ----
    {
      int nn = lane & 15;
      #pragma unroll
      for (int ct = 0; ct < 2; ++ct) {
        int crow = cblk + ct*16 + (lane & 15);
        int klo  = (lane < 16) ? 0 : 16;
        v16bf vf = cat8(*(const v8bf*)&sV[cur][crow][klo],
                        *(const v8bf*)&sV[cur][crow][klo + 8]);  // A[c][j]
        #pragma unroll
        for (int it = 0; it < 4; ++it) {
          int idx = ct*4 + it;
          float scale = sScale[it*16 + nn];
          #pragma unroll
          for (int r = 0; r < 8; ++r) o[idx][r] *= scale;
          int irow = it*16 + (lane & 15);
          int joff = (lane < 16) ? 0 : 16;
          v16bf pf = *(const v16bf*)&sP[irow][joff];        // B[j][i] = P[i][j]
          o[idx] = __builtin_amdgcn_wmma_f32_16x16x32_bf16(false, vf, false, pf,
                                                           (short)0, o[idx],
                                                           false, false);
        }
      }
    }
    // no end-of-loop barrier needed: next iteration's top wait+barrier
    // separates these sP/sV[cur] reads from the next tile's writes.
  }

  // ---- epilogue: O / l_i + residual x, channel-major f32 store ----
  {
    int nn = lane & 15;
    int mb = 8*(lane >> 4);
    const float* xb = x   + (size_t)b * C_DIM * N_TOK;
    float*       ob = out + (size_t)b * C_DIM * N_TOK;
    #pragma unroll
    for (int ct = 0; ct < 2; ++ct) {
      #pragma unroll
      for (int it = 0; it < 4; ++it) {
        int   idx = ct*4 + it;
        int   ig  = i0 + it*16 + nn;
        float inv = 1.0f / sL[it*16 + nn];
        #pragma unroll
        for (int r = 0; r < 8; ++r) {
          int c = cblk + ct*16 + mb + r;
          ob[(size_t)c*N_TOK + ig] = o[idx][r]*inv + xb[(size_t)c*N_TOK + ig];
        }
      }
    }
  }
}

// ===========================================================================
// Projection GEMMs  Y[M, t] = Wmat[M, C] * x[b][C, t] + bias
//   y=0 -> Qb[b][t][32], y=1 -> Kb[b][t][32] (token-major)
//   y=2..9 -> Vb[b][c][t] rows (y-2)*32 (channel-major, via LDS re-tile)
// Block: 256 thr (8 waves); block tile 32 M x 64 tokens; wave = one 16x16.
// ===========================================================================
__global__ __launch_bounds__(256)
void proj_kernel(const float* __restrict__ x,
                 const float* __restrict__ Wq, const float* __restrict__ bq,
                 const float* __restrict__ Wk, const float* __restrict__ bk,
                 const float* __restrict__ Wv, const float* __restrict__ bv,
                 bf16_t* __restrict__ Qb, bf16_t* __restrict__ Kb,
                 bf16_t* __restrict__ Vb) {
  __shared__ bf16_t sA[32][40];   // W chunk [m][k]     (80B stride, 16B align)
  __shared__ bf16_t sB[64][48];   // x chunk [token][k] (96B stride)
  __shared__ bf16_t sC[32][72];   // V out re-tile [m][token] (144B stride)

  const int tid  = threadIdx.x;
  const int lane = tid & 31;
  const int wave = tid >> 5;
  const int b    = blockIdx.z;
  const int t0   = blockIdx.x * 64;
  const int yb   = blockIdx.y;

  const float* Wm; const float* bias; bf16_t* outp; int m_base;
  if (yb == 0)      { Wm = Wq; bias = bq; outp = Qb + (size_t)b*N_TOK*CQ;    m_base = 0;         }
  else if (yb == 1) { Wm = Wk; bias = bk; outp = Kb + (size_t)b*N_TOK*CQ;    m_base = 0;         }
  else              { Wm = Wv; bias = bv; outp = Vb + (size_t)b*N_TOK*C_DIM; m_base = (yb-2)*32; }

  const float* xb = x + (size_t)b * C_DIM * N_TOK;

  const int mt = wave & 1;    // 16-row m tile (of 32)
  const int tt = wave >> 1;   // 16-token tile (of 64)

  v8f acc = {};

  for (int k0 = 0; k0 < C_DIM; k0 += 32) {
    // ---- stage A: W[m_base..+32][k0..+32] -> bf16 LDS (packed 8B stores) --
    {
      int e = tid * 4;                  // 1024 elements / 256 threads
      int r = e >> 5, c = e & 31;
      const float4 w4 = *(const float4*)&Wm[(size_t)(m_base + r)*C_DIM + k0 + c];
      v4bf a4;
      a4[0] = (bf16_t)w4.x; a4[1] = (bf16_t)w4.y;
      a4[2] = (bf16_t)w4.z; a4[3] = (bf16_t)w4.w;
      *(v4bf*)&sA[r][c] = a4;
    }
    // ---- stage B: x[k0..+32][t0..+64] -> LDS [t][k], packed-pair stores --
    {
      int kp = tid >> 4;                // 0..15 -> k = 2kp, 2kp+1
      int tg = (tid & 15) * 4;          // token group of 4
      float4 r0 = *(const float4*)&xb[(size_t)(k0 + 2*kp + 0)*N_TOK + t0 + tg];
      float4 r1 = *(const float4*)&xb[(size_t)(k0 + 2*kp + 1)*N_TOK + t0 + tg];
      const float a0[4] = {r0.x, r0.y, r0.z, r0.w};
      const float a1[4] = {r1.x, r1.y, r1.z, r1.w};
      #pragma unroll
      for (int i = 0; i < 4; ++i) {
        v2bf pp; pp[0] = (bf16_t)a0[i]; pp[1] = (bf16_t)a1[i];
        *(v2bf*)&sB[tg + i][2*kp] = pp;
      }
    }
    __syncthreads();

    // ---- WMMA fragments, contiguous k = 16h + s in both A and B (16-bit A 16x32, B 32x16) ----
    {
      int row = mt*16 + (lane & 15);
      int klo = (lane < 16) ? 0 : 16;
      v16bf afrag = cat8(*(const v8bf*)&sA[row][klo],
                         *(const v8bf*)&sA[row][klo + 8]);
      int tok  = tt*16 + (lane & 15);
      int koff = (lane < 16) ? 0 : 16;
      v16bf bfrag = *(const v16bf*)&sB[tok][koff];
      acc = __builtin_amdgcn_wmma_f32_16x16x32_bf16(false, afrag, false, bfrag,
                                                    (short)0, acc, false, false);
    }
    __syncthreads();
  }

  // ---- epilogue ----
  const int n    = lane & 15;
  const int mrow = mt*16 + 8*(lane >> 4);
  if (yb < 2) {
    // Q/K: token-major [t][32]; lane stores 8 contiguous channels (16B)
    int tokg = t0 + tt*16 + n;
    v8bf res;
    #pragma unroll
    for (int r = 0; r < 8; ++r)
      res[r] = (bf16_t)(acc[r] + bias[mrow + r]);
    *(v8bf*)&outp[(size_t)tokg * CQ + mrow] = res;
  } else {
    // V: re-tile through LDS so global stores are coalesced channel-major
    #pragma unroll
    for (int r = 0; r < 8; ++r)
      sC[mrow + r][tt*16 + n] = (bf16_t)(acc[r] + bias[m_base + mrow + r]);
    __syncthreads();
    int rr = tid >> 3;            // 0..31 local channel row
    int jj = (tid & 7) * 8;       // 0..56 token offset
    v8bf vrow = *(const v8bf*)&sC[rr][jj];
    *(v8bf*)&outp[(size_t)(m_base + rr)*N_TOK + t0 + jj] = vrow;
  }
}

// ===========================================================================
extern "C" void kernel_launch(void* const* d_in, const int* in_sizes, int n_in,
                              void* d_out, int out_size, void* d_ws, size_t ws_size,
                              hipStream_t stream) {
  const float* x  = (const float*)d_in[0];
  const float* Wq = (const float*)d_in[1];
  const float* bq = (const float*)d_in[2];
  const float* Wk = (const float*)d_in[3];
  const float* bk = (const float*)d_in[4];
  const float* Wv = (const float*)d_in[5];
  const float* bv = (const float*)d_in[6];
  float* out = (float*)d_out;

  const size_t qkBytes = (size_t)4 * N_TOK * CQ * sizeof(bf16_t);  // 1 MB each
  char* ws = (char*)d_ws;
  bf16_t* Qb = (bf16_t*)(ws);
  bf16_t* Kb = (bf16_t*)(ws + qkBytes);
  bf16_t* Vb = (bf16_t*)(ws + 2 * qkBytes);                        // 8 MB

  dim3 gp(N_TOK/64, 10, 4);   // 10 m-blocks: Q(1) + K(1) + V(8)
  proj_kernel<<<gp, 256, 0, stream>>>(x, Wq, bq, Wk, bk, Wv, bv, Qb, Kb, Vb);

  dim3 ga(N_TOK/64, 4);
  attn_kernel<<<ga, 256, 0, stream>>>(x, Qb, Kb, Vb, out);
}
